// Processor_43233140801768
// MI455X (gfx1250) — matrix operand order repaired and checked by reading, not compiled, not run on hardware
//
#include <hip/hip_runtime.h>
#include <hip/hip_bf16.h>

// ---------------- problem constants (match reference) ----------------
#define NN 50000
#define NE 250000
#define DD 128      // node feature dim == edge dim == hidden dim
#define NB 9        // blocks
#define LN_EPS 1e-5f

// ---------------- tile / LDS layout constants ----------------
#define TILE_M 128         // rows (edges or nodes) per workgroup; 4 waves x 32 rows
#define LDA 144            // bf16 activation row stride (128 + 16 pad, keeps 32B align, spreads banks)
#define LDW 144            // bf16 transposed-weight row stride (per output col: 128 K + pad)
#define LDO 132            // f32 output staging row stride

// byte offsets inside dynamic LDS
#define OFF_W   0                          // sWt : 128 cols * LDW bf16    = 36864 B
#define OFF_A   36864                      // sA  : 128 rows * LDA bf16    = 36864 B
#define OFF_H0  73728                      // sH0 : 128 * LDA bf16         = 36864 B
#define OFF_H1  110592                     // sH1 : 128 * LDA bf16         = 36864 B
#define OFF_MU  147456                     // 128 f32
#define OFF_RS  147968                     // 128 f32
#define LDS_BYTES 148480
// sO (f32, 128*LDO*4 = 67584 B) overlaps OFF_W+OFF_A regions, used only after layer-3 GEMM (barrier'd)

typedef __bf16 bf16_t;
typedef __attribute__((ext_vector_type(16))) __bf16 v16bf;
typedef __attribute__((ext_vector_type(8)))  __bf16 v8bf;
typedef __attribute__((ext_vector_type(8)))  float  v8f;

__device__ __forceinline__ bf16_t f2bf(float f) {
    union { float f; unsigned u; } x; x.f = f;
    unsigned r = x.u + 0x7FFFu + ((x.u >> 16) & 1u);   // round-to-nearest-even
    unsigned short h = (unsigned short)(r >> 16);
    return __builtin_bit_cast(bf16_t, h);
}

__device__ __forceinline__ v8f vzero8() {
    v8f v;
#pragma unroll
    for (int i = 0; i < 8; ++i) v[i] = 0.0f;
    return v;
}

// Load a 128x128 weight chunk (row-major K x N, f32) into LDS transposed bf16: sWt[n*LDW + k]
__device__ __forceinline__ void load_weights(const float* __restrict__ w,
                                             bf16_t* __restrict__ sWt, int t) {
    for (int idx = t; idx < 128 * 128; idx += 128) {
        int k = idx >> 7, n = idx & 127;               // n is fast dim -> coalesced global reads
        sWt[n * LDW + k] = f2bf(w[idx]);
    }
}

// Build a 16x32 A fragment in contiguous k order (same order as the B fragment):
// lane m=lane&15; lanes<16 carry K{0..15}, lanes>=16 carry K{16..31}
__device__ __forceinline__ v16bf load_a_frag(const bf16_t* __restrict__ arow, int kHalf) {
    v8bf a_lo = *(const v8bf*)(arow + kHalf);          // 16B aligned
    v8bf a_hi = *(const v8bf*)(arow + kHalf + 8);      // 16B aligned
    v16bf a;
#pragma unroll
    for (int i = 0; i < 8; ++i) { a[i] = a_lo[i]; a[8 + i] = a_hi[i]; }
    return a;
}

// One wave computes a 32x128 fp32 tile: two 16-row sub-tiles sharing every B fragment.
// acc[0..7]  = rows [waveRow,    waveRow+16) x cols [nt*16, nt*16+16)
// acc[8..15] = rows [waveRow+16, waveRow+32)
// B fragment: col = nt*16 + m; element e -> K = (lane<16?0:16)+e (contiguous in transposed LDS).
// Each B load now feeds two back-to-back WMMAs -> half the ds/wait per wmma.
__device__ __forceinline__ void gemm_tile_acc(const bf16_t* __restrict__ sAct,
                                              const bf16_t* __restrict__ sWt,
                                              int waveRow, int lane, v8f acc[16]) {
    const int m      = lane & 15;
    const int kHalf  = (lane < 16) ? 0 : 16;
    const int kBaseB = (lane < 16) ? 0 : 16;
    const bf16_t* arow0 = sAct + (waveRow + m) * LDA;
    const bf16_t* arow1 = sAct + (waveRow + 16 + m) * LDA;
    const bf16_t* bBase = sWt + m * LDW + kBaseB;
#pragma unroll
    for (int kt = 0; kt < 4; ++kt) {                   // K chunk = 128 = 4 * 32
        v16bf a0 = load_a_frag(arow0 + kt * 32, kHalf);
        v16bf a1 = load_a_frag(arow1 + kt * 32, kHalf);
#pragma unroll
        for (int nt = 0; nt < 8; ++nt) {
            v16bf b = *(const v16bf*)(bBase + nt * 16 * LDW + kt * 32); // 32B aligned
            acc[nt]     = __builtin_amdgcn_wmma_f32_16x16x32_bf16(
                false, a0, false, b, (short)0, acc[nt],     false, false);
            acc[8 + nt] = __builtin_amdgcn_wmma_f32_16x16x32_bf16(
                false, a1, false, b, (short)0, acc[8 + nt], false, false);
        }
    }
}

// C/D layout: VGPR j, lanes<16 -> row j, lanes>=16 -> row j+8; col = nt*16 + (lane&15)
__device__ __forceinline__ void store_half_bf16(bf16_t* __restrict__ dst, int rowBase, int lane,
                                                const v8f* acc, const float* __restrict__ bias,
                                                bool relu) {
    const int n0 = lane & 15;
    const int mOff = (lane < 16) ? 0 : 8;
#pragma unroll
    for (int nt = 0; nt < 8; ++nt) {
        int col = nt * 16 + n0;
        float bcol = bias[col];
#pragma unroll
        for (int j = 0; j < 8; ++j) {
            float v = acc[nt][j] + bcol;
            if (relu) v = fmaxf(v, 0.0f);
            dst[(rowBase + mOff + j) * LDA + col] = f2bf(v);
        }
    }
}

__device__ __forceinline__ void store_half_f32(float* __restrict__ dst, int rowBase, int lane,
                                               const v8f* acc, const float* __restrict__ bias) {
    const int n0 = lane & 15;
    const int mOff = (lane < 16) ? 0 : 8;
#pragma unroll
    for (int nt = 0; nt < 8; ++nt) {
        int col = nt * 16 + n0;
        float bcol = bias[col];
#pragma unroll
        for (int j = 0; j < 8; ++j)
            dst[(rowBase + mOff + j) * LDO + col] = acc[nt][j] + bcol;
    }
}

// LayerNorm stats: one thread per row (128 rows, 128 threads, all active)
__device__ __forceinline__ void ln_stats(const float* __restrict__ sO,
                                         float* __restrict__ sMu, float* __restrict__ sRs, int t) {
    const float* ro = sO + t * LDO;
    float s = 0.f, s2 = 0.f;
    for (int c = 0; c < 128; ++c) { float v = ro[c]; s += v; s2 += v * v; }
    float mu = s * (1.0f / 128.0f);
    float var = s2 * (1.0f / 128.0f) - mu * mu;
    sMu[t] = mu;
    sRs[t] = rsqrtf(var + LN_EPS);
    __syncthreads();
}

// ================= edge kernel: fused edge-MLP + LN + residual + atomic scatter =================
__global__ __launch_bounds__(128) void edge_kernel(
    const float* __restrict__ xin, const long long* __restrict__ src,
    const long long* __restrict__ dst, const float* __restrict__ eain,
    const float* __restrict__ w1, const float* __restrict__ b1,
    const float* __restrict__ w2, const float* __restrict__ b2,
    const float* __restrict__ w3, const float* __restrict__ b3,
    const float* __restrict__ g,  const float* __restrict__ bt,
    float* __restrict__ eaout, float* __restrict__ agg, int nE)
{
    extern __shared__ char lds[];
    bf16_t* sWt = (bf16_t*)(lds + OFF_W);
    bf16_t* sA  = (bf16_t*)(lds + OFF_A);
    bf16_t* sH0 = (bf16_t*)(lds + OFF_H0);
    bf16_t* sH1 = (bf16_t*)(lds + OFF_H1);
    float*  sO  = (float*)(lds + OFF_W);       // reused post-barrier (covers sWt + part of sA)
    float*  sMu = (float*)(lds + OFF_MU);
    float*  sRs = (float*)(lds + OFF_RS);

    const int t = threadIdx.x;
    const int lane = t & 31;
    const int waveRow = (t >> 5) * 32;
    const long base = (long)blockIdx.x * TILE_M;

    v8f acc[16];
#pragma unroll
    for (int i = 0; i < 16; ++i) acc[i] = vzero8();

    // ---- layer 1: K = 384 = 3 chunks of 128 (x[src] | x[dst] | ea) ----
    for (int c = 0; c < 3; ++c) {
        __syncthreads();
        load_weights(w1 + (long)c * 128 * 128, sWt, t);
        if (c < 2) {
            const long long* idx = (c == 0) ? src : dst;
            for (int q = t; q < TILE_M * 128; q += 128) {
                int r = q >> 7, col = q & 127;
                long e = base + r;
                long node = (e < nE) ? (long)idx[e] : 0;
                sA[r * LDA + col] = f2bf(xin[node * DD + col]);
            }
        } else {
            for (int q = t; q < TILE_M * 128; q += 128) {
                int r = q >> 7, col = q & 127;
                long e = base + r;
                sA[r * LDA + col] = f2bf((e < nE) ? eain[e * DD + col] : 0.0f);
            }
        }
        __syncthreads();
        gemm_tile_acc(sA, sWt, waveRow, lane, acc);
    }
    store_half_bf16(sH0, waveRow,      lane, acc,     b1, true);
    store_half_bf16(sH0, waveRow + 16, lane, acc + 8, b1, true);

    // ---- layer 2 ----
#pragma unroll
    for (int i = 0; i < 16; ++i) acc[i] = vzero8();
    __syncthreads();
    load_weights(w2, sWt, t);
    __syncthreads();
    gemm_tile_acc(sH0, sWt, waveRow, lane, acc);
    store_half_bf16(sH1, waveRow,      lane, acc,     b2, true);
    store_half_bf16(sH1, waveRow + 16, lane, acc + 8, b2, true);

    // ---- layer 3 ----
#pragma unroll
    for (int i = 0; i < 16; ++i) acc[i] = vzero8();
    __syncthreads();
    load_weights(w3, sWt, t);
    __syncthreads();
    gemm_tile_acc(sH1, sWt, waveRow, lane, acc);
    __syncthreads();                               // done reading sWt/sA -> safe to overwrite with sO
    store_half_f32(sO, waveRow,      lane, acc,     b3);
    store_half_f32(sO, waveRow + 16, lane, acc + 8, b3);
    __syncthreads();

    ln_stats(sO, sMu, sRs, t);

    // ---- epilogue: LN affine, residuals, atomic scatter-sum to agg[dst] ----
    {
        const int col = t;                         // 0..127, coalesced across threads
        const float gc = g[col], btc = bt[col];
        for (int r = 0; r < TILE_M; ++r) {
            long e = base + r;
            if (e >= nE) break;
            float v   = (sO[r * LDO + col] - sMu[r]) * sRs[r] * gc + btc;
            float eav = eain[e * DD + col];
            float mid = v + eav;                   // edge-processor output (feeds aggregation)
            eaout[e * DD + col] = mid + eav;       // + outer residual
            atomicAdd(&agg[(long)dst[e] * DD + col], mid);
        }
    }
}

// ================= node kernel: fused node-MLP + LN + residuals =================
__global__ __launch_bounds__(128) void node_kernel(
    const float* __restrict__ xin, const float* __restrict__ agg,
    const float* __restrict__ w1, const float* __restrict__ b1,
    const float* __restrict__ w2, const float* __restrict__ b2,
    const float* __restrict__ w3, const float* __restrict__ b3,
    const float* __restrict__ g,  const float* __restrict__ bt,
    float* __restrict__ xout, int nN)
{
    extern __shared__ char lds[];
    bf16_t* sWt = (bf16_t*)(lds + OFF_W);
    bf16_t* sA  = (bf16_t*)(lds + OFF_A);
    bf16_t* sH0 = (bf16_t*)(lds + OFF_H0);
    bf16_t* sH1 = (bf16_t*)(lds + OFF_H1);
    float*  sO  = (float*)(lds + OFF_W);
    float*  sMu = (float*)(lds + OFF_MU);
    float*  sRs = (float*)(lds + OFF_RS);

    const int t = threadIdx.x;
    const int lane = t & 31;
    const int waveRow = (t >> 5) * 32;
    const long base = (long)blockIdx.x * TILE_M;

    v8f acc[16];
#pragma unroll
    for (int i = 0; i < 16; ++i) acc[i] = vzero8();

    // ---- layer 1: K = 256 = 2 chunks of 128 (x | agg) ----
    for (int c = 0; c < 2; ++c) {
        __syncthreads();
        load_weights(w1 + (long)c * 128 * 128, sWt, t);
        const float* srcp = (c == 0) ? xin : agg;
        for (int q = t; q < TILE_M * 128; q += 128) {
            int r = q >> 7, col = q & 127;
            long n = base + r;
            sA[r * LDA + col] = f2bf((n < nN) ? srcp[n * DD + col] : 0.0f);
        }
        __syncthreads();
        gemm_tile_acc(sA, sWt, waveRow, lane, acc);
    }
    store_half_bf16(sH0, waveRow,      lane, acc,     b1, true);
    store_half_bf16(sH0, waveRow + 16, lane, acc + 8, b1, true);

    // ---- layer 2 ----
#pragma unroll
    for (int i = 0; i < 16; ++i) acc[i] = vzero8();
    __syncthreads();
    load_weights(w2, sWt, t);
    __syncthreads();
    gemm_tile_acc(sH0, sWt, waveRow, lane, acc);
    store_half_bf16(sH1, waveRow,      lane, acc,     b2, true);
    store_half_bf16(sH1, waveRow + 16, lane, acc + 8, b2, true);

    // ---- layer 3 ----
#pragma unroll
    for (int i = 0; i < 16; ++i) acc[i] = vzero8();
    __syncthreads();
    load_weights(w3, sWt, t);
    __syncthreads();
    gemm_tile_acc(sH1, sWt, waveRow, lane, acc);
    __syncthreads();
    store_half_f32(sO, waveRow,      lane, acc,     b3);
    store_half_f32(sO, waveRow + 16, lane, acc + 8, b3);
    __syncthreads();

    ln_stats(sO, sMu, sRs, t);

    {
        const int col = t;
        const float gc = g[col], btc = bt[col];
        for (int r = 0; r < TILE_M; ++r) {
            long n = base + r;
            if (n >= nN) break;
            float v = (sO[r * LDO + col] - sMu[r]) * sRs[r] * gc + btc;
            // node residual + outer block residual: x_next = mlp + 2*x
            xout[n * DD + col] = v + 2.0f * xin[n * DD + col];
        }
    }
}

__global__ void zero_kernel(float* __restrict__ p, long n) {
    long i = (long)blockIdx.x * blockDim.x + threadIdx.x;
    if (i < n) p[i] = 0.0f;
}

// =========================== host launcher ===========================
extern "C" void kernel_launch(void* const* d_in, const int* in_sizes, int n_in,
                              void* d_out, int out_size, void* d_ws, size_t ws_size,
                              hipStream_t stream) {
    const float*     x    = (const float*)d_in[0];
    const long long* eidx = (const long long*)d_in[1];     // int64 per reference
    const float*     ea   = (const float*)d_in[2];
    const float* ew1 = (const float*)d_in[3];  const float* eb1 = (const float*)d_in[4];
    const float* ew2 = (const float*)d_in[5];  const float* eb2 = (const float*)d_in[6];
    const float* ew3 = (const float*)d_in[7];  const float* eb3 = (const float*)d_in[8];
    const float* eg  = (const float*)d_in[9];  const float* ebt = (const float*)d_in[10];
    const float* nw1 = (const float*)d_in[11]; const float* nb1 = (const float*)d_in[12];
    const float* nw2 = (const float*)d_in[13]; const float* nb2 = (const float*)d_in[14];
    const float* nw3 = (const float*)d_in[15]; const float* nb3 = (const float*)d_in[16];
    const float* ng  = (const float*)d_in[17]; const float* nbt = (const float*)d_in[18];

    const long long* src = eidx;            // edge_index[0]
    const long long* dst = eidx + NE;       // edge_index[1]

    // workspace layout (floats)
    float* ws   = (float*)d_ws;
    float* xA   = ws;                               // 50000*128
    float* eaA  = xA  + (long)NN * DD;              // 250000*128
    float* eaB  = eaA + (long)NE * DD;              // 250000*128
    float* agg  = eaB + (long)NE * DD;              // 50000*128

    // allow >64KB dynamic LDS
    (void)hipFuncSetAttribute((const void*)edge_kernel,
                              hipFuncAttributeMaxDynamicSharedMemorySize, LDS_BYTES);
    (void)hipFuncSetAttribute((const void*)node_kernel,
                              hipFuncAttributeMaxDynamicSharedMemorySize, LDS_BYTES);

    const int edgeBlocks = (NE + TILE_M - 1) / TILE_M;   // 1954
    const int nodeBlocks = (NN + TILE_M - 1) / TILE_M;   // 391
    const long aggN = (long)NN * DD;

    const float* xc  = x;
    const float* eac = ea;
    float* xbuf[2]  = {(float*)d_out, xA};               // block 8 (even) lands in d_out
    float* eabuf[2] = {eaA, eaB};

    for (int b = 0; b < NB; ++b) {
        float* xn  = xbuf[b & 1];
        float* ean = eabuf[b & 1];

        zero_kernel<<<(int)((aggN + 255) / 256), 256, 0, stream>>>(agg, aggN);

        edge_kernel<<<edgeBlocks, 128, LDS_BYTES, stream>>>(
            xc, src, dst, eac,
            ew1 + (long)b * (3 * 128) * 128, eb1 + b * 128,
            ew2 + (long)b * 128 * 128,       eb2 + b * 128,
            ew3 + (long)b * 128 * 128,       eb3 + b * 128,
            eg + b * 128, ebt + b * 128,
            ean, agg, NE);

        node_kernel<<<nodeBlocks, 128, LDS_BYTES, stream>>>(
            xc, agg,
            nw1 + (long)b * (2 * 128) * 128, nb1 + b * 128,
            nw2 + (long)b * 128 * 128,       nb2 + b * 128,
            nw3 + (long)b * 128 * 128,       nb3 + b * 128,
            ng + b * 128, nbt + b * 128,
            xn, NN);

        xc = xn;
        eac = ean;
    }
}
